// LinearSelfAttention_78065325572222
// MI455X (gfx1250) — compile-verified, matrix operand order checked by reading, not run on hardware
//
#include <hip/hip_runtime.h>

#define C_DIM 384
#define QKV_STRIDE 769
#define N_TOK 1024
#define TILE_M 64
#define XA_STRIDE 392   // 384 + 8 bf16 pad
#define WB_STRIDE 40    // 32 + 8 bf16 pad (matches TDM pad: 64B data + 16B pad per row)
#define WB_ELEMS (C_DIM * WB_STRIDE)   // one wb buffer, in bf16 elems

typedef __bf16 bf16;
typedef bf16 v16bf __attribute__((ext_vector_type(16)));
typedef bf16 v8bf  __attribute__((ext_vector_type(8)));
typedef float v8f  __attribute__((ext_vector_type(8)));
typedef unsigned int v4u __attribute__((ext_vector_type(4)));
typedef int v8i __attribute__((ext_vector_type(8)));
typedef int v4i __attribute__((ext_vector_type(4)));

__device__ __forceinline__ unsigned short f2bf_rne(float f) {
  unsigned int u = __float_as_uint(f);
  u += 0x7FFFu + ((u >> 16) & 1u);
  return (unsigned short)(u >> 16);
}

__device__ __forceinline__ v16bf frag16(const unsigned short* lo, const unsigned short* hi) {
  v8bf a = *(const v8bf*)lo;   // 16B-aligned LDS vector load
  v8bf b = *(const v8bf*)hi;
  v16bf r;
#pragma unroll
  for (int i = 0; i < 8; ++i) { r[i] = a[i]; r[i + 8] = b[i]; }
  return r;
}

// ---- TDM: DMA one 384x32 bf16 weight K-chunk global->LDS with 16B/row padding ----
// D# per CDNA5 ISA. 2-D tensor, groups 2/3 zero.
// This toolchain's builtin takes 6 args: (v4u g0, v8i g1, v4i g2, v4i g3, v8i g4, i32 cpol).
__device__ __forceinline__ void tdm_load_chunk(const unsigned short* wT, int kbase,
                                               unsigned lds_byte) {
  unsigned long long ga = (unsigned long long)(const void*)(wT + kbase);
  v4u g0;
  g0[0] = 1u;                                            // count=1, user mode
  g0[1] = lds_byte;                                      // lds_addr (bytes)
  g0[2] = (unsigned)(ga & 0xFFFFFFFFu);                  // global_addr[31:0]
  g0[3] = (unsigned)((ga >> 32) & 0x01FFFFFFu) | (2u << 30);  // addr[56:32] | type=2
  v8i g1;
  // data_size=1(2B) | pad_enable | pad_interval=3(16 DW=64B) | pad_amount=3(4 DW=16B)
  g1[0] = (int)((1u << 16) | (1u << 20) | (3u << 22) | (3u << 25));
  g1[1] = (int)(384u << 16);   // tensor_dim0[15:0]=384 in bits[79:48] low half
  g1[2] = (int)(384u << 16);   // tensor_dim0 hi=0 ; tensor_dim1[15:0]=384
  g1[3] = (int)(32u << 16);    // tensor_dim1 hi=0 ; tile_dim0=32 (K elems/row)
  g1[4] = (int)(384u);         // tile_dim1=384 rows ; tile_dim2=0
  g1[5] = (int)(384u);         // tensor_dim0_stride = 384 elems (row pitch)
  g1[6] = 0; g1[7] = 0;        // stride hi / tensor_dim1_stride = 0
  v4i g2 = {0, 0, 0, 0};
  v4i g3 = {0, 0, 0, 0};
  v8i g4 = {0, 0, 0, 0, 0, 0, 0, 0};
  __builtin_amdgcn_tensor_load_to_lds(g0, g1, g2, g3, g4, 0);
}

// ---------------- Kernel 0: weight convert + transpose to bf16 [col][k] ----------------
__global__ void wprep_kernel(const float* __restrict__ w_qkv,
                             const float* __restrict__ w_proj,
                             unsigned short* __restrict__ wvT,
                             unsigned short* __restrict__ wpT) {
  int k = blockIdx.x * 16 + threadIdx.x;   // K index (row of W)
  int c = blockIdx.y * 16 + threadIdx.y;   // output column
  if (k < C_DIM && c < C_DIM) {
    if (blockIdx.z == 0)
      wvT[c * C_DIM + k] = f2bf_rne(w_qkv[k * QKV_STRIDE + 1 + C_DIM + c]);  // value block
    else
      wpT[c * C_DIM + k] = f2bf_rne(w_proj[k * C_DIM + c]);
  }
}

// ---------------- Kernel 1: context vector (softmax over size-1 axis == 1.0) ----------------
__global__ void cv_kernel(const float* __restrict__ x,
                          const float* __restrict__ w_qkv,
                          const float* __restrict__ b_qkv,
                          float* __restrict__ cv) {
  __shared__ float xsum[C_DIM];
  const int bp = blockIdx.x;
  const int c = threadIdx.x;                 // 384 threads = 12 waves
  const float* xs = x + (size_t)bp * N_TOK * C_DIM + c;
  float s = 0.f;
  for (int n = 0; n < N_TOK; ++n) s += xs[(size_t)n * C_DIM];   // coalesced over c
  xsum[c] = s;
  __syncthreads();
  float acc = 0.f;
  for (int k = 0; k < C_DIM; ++k) acc += xsum[k] * w_qkv[k * QKV_STRIDE + 1 + c];
  cv[bp * C_DIM + c] = acc + (float)N_TOK * b_qkv[1 + c];
}

// ---------------- fused main kernel: one 384-K GEMM with TDM-double-buffered weights ----
__device__ __forceinline__ void gemm384(const unsigned short* __restrict__ wT,
                                        const unsigned short* __restrict__ xa,
                                        unsigned short* __restrict__ wb,   // 2 buffers
                                        unsigned wb_byte0,
                                        int ln, int h, int rt, int ch, bool issuer,
                                        v8f acc[12]) {
  // hoisted fragment base pointers: per-j / per-kc offsets are compile-time constants
  const unsigned short* abase = xa + (rt * 16 + ln) * XA_STRIDE + h * 8;
  const unsigned short* bbase = wb + (ch * 192 + ln) * WB_STRIDE + h * 8;

  __syncthreads();          // previous users of wb / writers of xa are done
  if (issuer) tdm_load_chunk(wT, 0, wb_byte0);              // prefetch chunk 0
#pragma unroll
  for (int kc = 0; kc < 12; ++kc) {
    const int kbase = kc * 32;
    if (issuer) {
      if (kc < 11) {
        tdm_load_chunk(wT, kbase + 32,
                       wb_byte0 + (unsigned)(((kc + 1) & 1) * WB_ELEMS * 2));
        __builtin_amdgcn_s_wait_tensorcnt(1);   // chunk kc landed, kc+1 in flight
      } else {
        __builtin_amdgcn_s_wait_tensorcnt(0);
      }
    }
    __syncthreads();        // publish buffer kc&1 to all waves
    const unsigned short* bcur = bbase + (kc & 1) * WB_ELEMS;
    v16bf afrag = frag16(abase + kbase, abase + kbase + 16);
#pragma unroll
    for (int j = 0; j < 12; ++j) {
      const unsigned short* bcol = bcur + j * (16 * WB_STRIDE);   // +1280 B steps
      v16bf bfrag = frag16(bcol, bcol + 16);
      acc[j] = __builtin_amdgcn_wmma_f32_16x16x32_bf16(
          false, afrag, false, bfrag, (short)0, acc[j], false, false);
    }
    if (kc < 11) __syncthreads();   // readers done before buffer kc&1 is re-DMA'd
  }
}

__global__ void __launch_bounds__(256)
fused_kernel(const float* __restrict__ x,
             const unsigned short* __restrict__ wvT,
             const unsigned short* __restrict__ wpT,
             const float* __restrict__ b_qkv,
             const float* __restrict__ b_proj,
             const float* __restrict__ cv,
             float* __restrict__ out) {
  extern __shared__ unsigned short smem[];
  unsigned short* xa = smem;                                // [64][392] bf16: x tile -> gated tile
  unsigned short* wb = smem + TILE_M * XA_STRIDE;           // 2 x [384][40] bf16 TDM buffers
  float* cvs = (float*)(smem + TILE_M * XA_STRIDE + 2 * WB_ELEMS);
  float* bvs = cvs + C_DIM;
  float* bps = bvs + C_DIM;
  const unsigned wb_byte0 =
      (unsigned)(unsigned long long)(const void*)wb;        // LDS aperture: low 32 bits = offset

  const int t = threadIdx.x;
  const int lane = t & 31, wv = t >> 5;
  const int ln = lane & 15, h = lane >> 4;
  const int rt = wv & 3;          // row tile (16 rows)
  const int ch = wv >> 2;         // column half (192 cols)
  const bool issuer = (wv == 0);
  const int bp = blockIdx.x >> 4;                           // 16 token-tiles of 64 per (b,p)
  const size_t tok0 = (size_t)bp * N_TOK + (size_t)(blockIdx.x & 15) * TILE_M;

  // stage cv, biases
  for (int c = t; c < C_DIM; c += 256) {
    cvs[c] = cv[bp * C_DIM + c];
    bvs[c] = b_qkv[1 + C_DIM + c];
    bps[c] = b_proj[c];
  }
  // stage x tile -> bf16 LDS (coalesced 16B loads, RNE convert)
  const float* xbase = x + tok0 * C_DIM;
#pragma unroll
  for (int i = 0; i < 24; ++i) {
    int j = i * 256 + t;                 // float4 id; 64*96 total
    int r = j / 96, c4 = (j % 96) * 4;
    float4 f = *(const float4*)(xbase + (size_t)r * C_DIM + c4);
    uint2 p;
    p.x = (unsigned)f2bf_rne(f.x) | ((unsigned)f2bf_rne(f.y) << 16);
    p.y = (unsigned)f2bf_rne(f.z) | ((unsigned)f2bf_rne(f.w) << 16);
    *(uint2*)(xa + r * XA_STRIDE + c4) = p;
  }

  v8f acc[12];
#pragma unroll
  for (int j = 0; j < 12; ++j)
#pragma unroll
    for (int e = 0; e < 8; ++e) acc[j][e] = 0.f;

  // GEMM1: v = x @ Wv   (leading barrier inside also covers the staging above)
  gemm384(wvT, xa, wb, wb_byte0, ln, h, rt, ch, issuer, acc);

  __syncthreads();   // all waves done reading xa -> safe to overwrite with gated tile

  // epilogue 1: gated = relu(v + bv) * cv  -> bf16 back into xa
#pragma unroll
  for (int j = 0; j < 12; ++j) {
    int col = ch * 192 + j * 16 + ln;
    float cvc = cvs[col], bvc = bvs[col];
#pragma unroll
    for (int i = 0; i < 8; ++i) {
      int row = rt * 16 + h * 8 + i;     // 16x16 f32 D layout: M = i + 8*h
      float g = fmaxf(acc[j][i] + bvc, 0.f) * cvc;
      xa[row * XA_STRIDE + col] = f2bf_rne(g);
    }
  }

#pragma unroll
  for (int j = 0; j < 12; ++j)
#pragma unroll
    for (int e = 0; e < 8; ++e) acc[j][e] = 0.f;

  // GEMM2: out = gated @ Wp   (leading barrier orders gated writes vs wb DMA & xa reads)
  gemm384(wpT, xa, wb, wb_byte0, ln, h, rt, ch, issuer, acc);

  // epilogue 2: + b_proj, store fp32
  float* obase = out + tok0 * C_DIM;
#pragma unroll
  for (int j = 0; j < 12; ++j) {
    int col = ch * 192 + j * 16 + ln;
    float bpc = bps[col];
#pragma unroll
    for (int i = 0; i < 8; ++i) {
      int row = rt * 16 + h * 8 + i;
      obase[(size_t)row * C_DIM + col] = acc[j][i] + bpc;
    }
  }
}

extern "C" void kernel_launch(void* const* d_in, const int* in_sizes, int n_in,
                              void* d_out, int out_size, void* d_ws, size_t ws_size,
                              hipStream_t stream) {
  (void)in_sizes; (void)n_in; (void)out_size; (void)ws_size;
  const float* x      = (const float*)d_in[0];
  const float* w_qkv  = (const float*)d_in[1];
  const float* b_qkv  = (const float*)d_in[2];
  const float* w_proj = (const float*)d_in[3];
  const float* b_proj = (const float*)d_in[4];
  float* out = (float*)d_out;

  // workspace layout: wvT bf16 [384*384] | wpT bf16 [384*384] | cv f32 [128*384]
  unsigned short* wvT = (unsigned short*)d_ws;
  unsigned short* wpT = wvT + C_DIM * C_DIM;
  float* cvw = (float*)(wpT + C_DIM * C_DIM);

  wprep_kernel<<<dim3(24, 24, 2), dim3(16, 16), 0, stream>>>(w_qkv, w_proj, wvT, wpT);
  cv_kernel<<<dim3(128), dim3(C_DIM), 0, stream>>>(x, w_qkv, b_qkv, cvw);

  size_t shmem = (size_t)(TILE_M * XA_STRIDE + 2 * WB_ELEMS) * sizeof(unsigned short)
               + 3 * C_DIM * sizeof(float);   // 116,224 B -> 2 blocks per WGP (320 KB LDS)
  fused_kernel<<<dim3(2048), dim3(256), shmem, stream>>>(x, wvT, wpT, b_qkv, b_proj, cvw, out);
}
